// TextSIRNN_89902255439929
// MI455X (gfx1250) — matrix operand order repaired and checked by reading, not compiled, not run on hardware
//
#include <hip/hip_runtime.h>

#define SEQ 200
#define BATCH 256
#define EMBED 300
#define EMB_PAD 320
#define HID 256
#define G4 1024          // 4*HID
#define NOUT 10

typedef __attribute__((ext_vector_type(16))) _Float16 v16h;
typedef __attribute__((ext_vector_type(8)))  _Float16 v8h;
typedef __attribute__((ext_vector_type(8)))  float    v8f;

// ---------------------------------------------------------------------------
// WMMA fragment loaders (row-major source, 'stride' elements per row).
// A 16x32 f16 (M x K): lanes 0-15 rows M=0..15 hold K 0-7 & 16-23;
//                      lanes 16-31 same rows hold K 8-15 & 24-31.
__device__ __forceinline__ v16h load_fragA(const _Float16* base, int stride, int lane) {
    int r = lane & 15;
    int khalf = (lane & 16) ? 8 : 0;
    const _Float16* p = base + (size_t)r * stride + khalf;
    v8h lo = *(const v8h*)(p);        // K  +0..7
    v8h hi = *(const v8h*)(p + 16);   // K +16..23
    v16h o;
#pragma unroll
    for (int i = 0; i < 8; ++i) { o[i] = lo[i]; o[8 + i] = hi[i]; }
    return o;
}

// B 32x16 f16 (K x N), stored as W[n][k] row-major:
// lanes 0-15 (cols N=0..15) hold K 0-7 & 16-23; lanes 16-31 hold K 8-15 & 24-31.
__device__ __forceinline__ v16h load_fragB(const _Float16* base, int stride, int lane) {
    int r = lane & 15;
    int koff = (lane & 16) ? 8 : 0;
    const _Float16* p = base + (size_t)r * stride + koff;
    v8h lo = *(const v8h*)(p);        // K +0..7
    v8h hi = *(const v8h*)(p + 16);   // K +16..23
    v16h o;
#pragma unroll
    for (int i = 0; i < 8; ++i) { o[i] = lo[i]; o[8 + i] = hi[i]; }
    return o;
}

__device__ __forceinline__ float sigf(float x) { return 1.f / (1.f + __expf(-x)); }
// Fast tanh via hardware exp: (e^{2x}-1)/(e^{2x}+1); saturates correctly.
__device__ __forceinline__ float tanh_fast(float x) {
    float e = __expf(2.f * x);
    return (e - 1.f) / (e + 1.f);
}

// ---------------------------------------------------------------------------
// Embedding gather + cast to f16, K padded 300 -> 320 with zeros.
__global__ void embed_kernel(const int* __restrict__ x, const float* __restrict__ emb,
                             _Float16* __restrict__ xs) {
    int tok = blockIdx.x;                       // SEQ*BATCH tokens
    int idx = x[tok];
    const float* src = emb + (size_t)idx * EMBED;
    _Float16* dst = xs + (size_t)tok * EMB_PAD;
    for (int k = threadIdx.x; k < EMB_PAD; k += blockDim.x)
        dst[k] = (k < EMBED) ? (_Float16)src[k] : (_Float16)0.f;
}

// Cast f32 weights (rows x ksrc) -> f16 (rows x kpad), zero-padded.
__global__ void castpad_kernel(const float* __restrict__ src, _Float16* __restrict__ dst,
                               int rows, int ksrc, int kpad) {
    size_t i = (size_t)blockIdx.x * blockDim.x + threadIdx.x;
    size_t total = (size_t)rows * kpad;
    if (i >= total) return;
    int r = (int)(i / kpad), k = (int)(i % kpad);
    dst[i] = (k < ksrc) ? (_Float16)src[(size_t)r * ksrc + k] : (_Float16)0.f;
}

__global__ void biassum_kernel(const float* __restrict__ a, const float* __restrict__ b,
                               float* __restrict__ o, int n) {
    int i = blockIdx.x * blockDim.x + threadIdx.x;
    if (i < n) o[i] = a[i] + b[i];
}

// ---------------------------------------------------------------------------
// Time-parallel input projection: out[m][n] = X[m][:] . W[n][:] + bias[n]
// X: (M, kpad) f16   W: (G4, kpad) f16   out: (M, G4) f32
// Block = 128 threads (4 waves); wave handles 4 N-tiles; grid (M/16, 4).
__global__ __launch_bounds__(128)
void gates_gemm_kernel(const _Float16* __restrict__ X, const _Float16* __restrict__ W,
                       const float* __restrict__ bias, float* __restrict__ out,
                       int kpad, int kiters) {
    int lane = threadIdx.x & 31;
    int wave = threadIdx.x >> 5;
    int m0 = blockIdx.x * 16;
    int nbase = blockIdx.y * 256 + wave * 64;

    v8f acc[4] = {};
    const _Float16* xrow = X + (size_t)m0 * kpad;
    for (int kb = 0; kb < kiters; ++kb) {
        v16h a = load_fragA(xrow + kb * 32, kpad, lane);
#pragma unroll
        for (int i = 0; i < 4; ++i) {
            v16h b = load_fragB(W + (size_t)(nbase + i * 16) * kpad + kb * 32, kpad, lane);
            acc[i] = __builtin_amdgcn_wmma_f32_16x16x32_f16(false, a, false, b,
                                                            (short)0, acc[i], false, false);
        }
    }
    int rbase = m0 + ((lane & 16) ? 8 : 0);
#pragma unroll
    for (int i = 0; i < 4; ++i) {
        int col = nbase + i * 16 + (lane & 15);
        float bv = bias[col];
#pragma unroll
        for (int v = 0; v < 8; ++v)
            out[(size_t)(rbase + v) * G4 + col] = acc[i][v] + bv;
    }
}

// ---------------------------------------------------------------------------
// Persistent recurrent kernel. Batch-row independence of the LSTM recurrence
// lets each workgroup own 16 batch rows for the full 200-step scan with only
// workgroup barriers. Grid (BATCH/16, 2 directions), block 512 (16 waves).
// Wave w owns hidden units [16w,16w+16): its 4 N-tiles are exactly the
// i,f,g,o gate columns for those units, so the LSTM cell update is purely
// per-lane arithmetic on the WMMA accumulators (no gate exchange).
__global__ __launch_bounds__(512)
void lstm_seq_kernel(const float* __restrict__ gx_all,     // (2, SEQ, BATCH, G4) f32
                     const _Float16* __restrict__ whh_all, // (2, G4, HID) f16
                     _Float16* __restrict__ ys,            // (SEQ, BATCH, 2*HID) or null
                     float* __restrict__ feat,             // (BATCH, 4*HID)
                     int feat_off, int write_ys) {
    __shared__ __align__(16) _Float16 hbuf[16 * HID];
    int bc = blockIdx.x;
    int d  = blockIdx.y;
    int lane = threadIdx.x & 31;
    int wave = threadIdx.x >> 5;                // 0..15
    const float*    gx  = gx_all  + (size_t)d * SEQ * BATCH * G4;
    const _Float16* whh = whh_all + (size_t)d * G4 * HID;

    for (int i = threadIdx.x; i < 16 * HID; i += blockDim.x) hbuf[i] = (_Float16)0.f;

    float creg[8], hreg[8];
#pragma unroll
    for (int i = 0; i < 8; ++i) { creg[i] = 0.f; hreg[i] = 0.f; }

    int j0   = wave * 16;                       // hidden-unit group for this wave
    int rlow = (lane & 16) ? 8 : 0;
    int jc   = lane & 15;
    int j    = j0 + jc;

    for (int step = 0; step < SEQ; ++step) {
        int t = d ? (SEQ - 1 - step) : step;
        __syncthreads();                        // hbuf holds h for this step
        v16h afrag[8];
#pragma unroll
        for (int kb = 0; kb < 8; ++kb)
            afrag[kb] = load_fragA(&hbuf[kb * 32], HID, lane);
        __syncthreads();                        // safe to overwrite hbuf below

        // Prefetch next step's gate pre-activation slab (16 rows x 4K bytes =
        // 64KB = 512 x 128B cachelines, one per thread) behind the WMMAs.
        {
            int stepn = (step + 1 < SEQ) ? step + 1 : step;
            int tn = d ? (SEQ - 1 - stepn) : stepn;
            const float* gnext = gx + ((size_t)tn * BATCH + bc * 16) * G4;
            __builtin_prefetch(gnext + (size_t)threadIdx.x * 32, 0, 0);
        }

        // h @ W_hh^T for this wave's 4 gate tiles. K-outer so the 4
        // accumulators form independent WMMA chains (pipelines loads).
        v8f acc[4] = {};
#pragma unroll
        for (int kb = 0; kb < 8; ++kb) {
#pragma unroll
            for (int g = 0; g < 4; ++g) {
                v16h b = load_fragB(whh + (size_t)(g * HID + j0) * HID + kb * 32, HID, lane);
                acc[g] = __builtin_amdgcn_wmma_f32_16x16x32_f16(false, afrag[kb], false, b,
                                                                (short)0, acc[g], false, false);
            }
        }

        // Element-wise LSTM cell (PyTorch gate order i,f,g,o)
        const float* grow = gx + ((size_t)t * BATCH + bc * 16) * G4;
#pragma unroll
        for (int v = 0; v < 8; ++v) {
            int m = rlow + v;
            const float* g = grow + (size_t)m * G4;
            float gi = acc[0][v] + g[j];
            float gf = acc[1][v] + g[HID + j];
            float gg = acc[2][v] + g[2 * HID + j];
            float go = acc[3][v] + g[3 * HID + j];
            float c = sigf(gf) * creg[v] + sigf(gi) * tanh_fast(gg);
            float h = sigf(go) * tanh_fast(c);
            creg[v] = c;
            hreg[v] = h;
            hbuf[m * HID + j] = (_Float16)h;
            if (write_ys)
                ys[((size_t)t * BATCH + bc * 16 + m) * (2 * HID) + d * HID + j] = (_Float16)h;
        }
    }
    // Final hidden state -> feature vector (order: l0f, l0b, l1f, l1b)
#pragma unroll
    for (int v = 0; v < 8; ++v) {
        int m = rlow + v;
        feat[(size_t)(bc * 16 + m) * (4 * HID) + feat_off + d * HID + j] = hreg[v];
    }
}

// ---------------------------------------------------------------------------
// FC(1024->10) + LayerNorm + log_softmax. One wave per batch row.
__global__ __launch_bounds__(32)
void head_kernel(const float* __restrict__ feat, const float* __restrict__ fcw,
                 const float* __restrict__ fcb, float* __restrict__ out) {
    int b = blockIdx.x;
    int lane = threadIdx.x;
    const float* f = feat + (size_t)b * (4 * HID);
    float acc[NOUT];
#pragma unroll
    for (int o = 0; o < NOUT; ++o) acc[o] = 0.f;
    for (int k = lane; k < 4 * HID; k += 32) {
        float fv = f[k];
#pragma unroll
        for (int o = 0; o < NOUT; ++o) acc[o] += fv * fcw[o * 4 * HID + k];
    }
#pragma unroll
    for (int o = 0; o < NOUT; ++o)
#pragma unroll
        for (int off = 16; off > 0; off >>= 1)
            acc[o] += __shfl_xor(acc[o], off, 32);
    if (lane == 0) {
        float v[NOUT];
        float mean = 0.f;
#pragma unroll
        for (int o = 0; o < NOUT; ++o) { v[o] = acc[o] + fcb[o]; mean += v[o]; }
        mean *= (1.f / NOUT);
        float var = 0.f;
#pragma unroll
        for (int o = 0; o < NOUT; ++o) { float dd = v[o] - mean; var += dd * dd; }
        var *= (1.f / NOUT);
        float inv = rsqrtf(var + 1e-5f);
        float mx = -1e30f;
#pragma unroll
        for (int o = 0; o < NOUT; ++o) { v[o] = (v[o] - mean) * inv; mx = fmaxf(mx, v[o]); }
        float s = 0.f;
#pragma unroll
        for (int o = 0; o < NOUT; ++o) s += __expf(v[o] - mx);
        float lse = mx + __logf(s);
#pragma unroll
        for (int o = 0; o < NOUT; ++o) out[(size_t)b * NOUT + o] = v[o] - lse;
    }
}

// ---------------------------------------------------------------------------
extern "C" void kernel_launch(void* const* d_in, const int* in_sizes, int n_in,
                              void* d_out, int out_size, void* d_ws, size_t ws_size,
                              hipStream_t stream) {
    (void)in_sizes; (void)n_in; (void)out_size; (void)ws_size;
    const int*   x   = (const int*)d_in[0];
    const float* emb = (const float*)d_in[1];
    const float* wih[2][2] = {{(const float*)d_in[2],  (const float*)d_in[6]},
                              {(const float*)d_in[10], (const float*)d_in[14]}};
    const float* whh[2][2] = {{(const float*)d_in[3],  (const float*)d_in[7]},
                              {(const float*)d_in[11], (const float*)d_in[15]}};
    const float* bih[2][2] = {{(const float*)d_in[4],  (const float*)d_in[8]},
                              {(const float*)d_in[12], (const float*)d_in[16]}};
    const float* bhh[2][2] = {{(const float*)d_in[5],  (const float*)d_in[9]},
                              {(const float*)d_in[13], (const float*)d_in[17]}};
    const float* fcw = (const float*)d_in[18];
    const float* fcb = (const float*)d_in[19];
    float* out = (float*)d_out;

    char* ws = (char*)d_ws;
    size_t off = 0;
    auto alloc = [&](size_t bytes) -> void* {
        void* p = (void*)(ws + off);
        off += (bytes + 255) & ~(size_t)255;
        return p;
    };
    _Float16* xs0   = (_Float16*)alloc((size_t)SEQ * BATCH * EMB_PAD * 2);
    _Float16* xs1   = (_Float16*)alloc((size_t)SEQ * BATCH * 2 * HID * 2);
    float*    gates = (float*)   alloc((size_t)2 * SEQ * BATCH * G4 * 4);
    _Float16* wih0h = (_Float16*)alloc((size_t)2 * G4 * EMB_PAD * 2);
    _Float16* wih1h = (_Float16*)alloc((size_t)2 * G4 * 2 * HID * 2);
    _Float16* whhh  = (_Float16*)alloc((size_t)4 * G4 * HID * 2);
    float*    biases= (float*)   alloc((size_t)4 * G4 * 4);
    float*    feat  = (float*)   alloc((size_t)BATCH * 4 * HID * 4);

    // Weight prep (f32 -> f16, K padded) + bias sums
    for (int d = 0; d < 2; ++d) {
        castpad_kernel<<<(G4 * EMB_PAD + 255) / 256, 256, 0, stream>>>(
            wih[0][d], wih0h + (size_t)d * G4 * EMB_PAD, G4, EMBED, EMB_PAD);
        castpad_kernel<<<(G4 * 2 * HID + 255) / 256, 256, 0, stream>>>(
            wih[1][d], wih1h + (size_t)d * G4 * 2 * HID, G4, 2 * HID, 2 * HID);
    }
    for (int l = 0; l < 2; ++l)
        for (int d = 0; d < 2; ++d) {
            castpad_kernel<<<(G4 * HID + 255) / 256, 256, 0, stream>>>(
                whh[l][d], whhh + (size_t)(l * 2 + d) * G4 * HID, G4, HID, HID);
            biassum_kernel<<<(G4 + 255) / 256, 256, 0, stream>>>(
                bih[l][d], bhh[l][d], biases + (size_t)(l * 2 + d) * G4, G4);
        }

    // Embedding gather
    embed_kernel<<<SEQ * BATCH, 64, 0, stream>>>(x, emb, xs0);

    const int M = SEQ * BATCH;
    // Layer 0: time-parallel input projections, then persistent recurrence
    for (int d = 0; d < 2; ++d)
        gates_gemm_kernel<<<dim3(M / 16, 4), 128, 0, stream>>>(
            xs0, wih0h + (size_t)d * G4 * EMB_PAD, biases + (size_t)d * G4,
            gates + (size_t)d * M * G4, EMB_PAD, EMB_PAD / 32);
    lstm_seq_kernel<<<dim3(BATCH / 16, 2), 512, 0, stream>>>(
        gates, whhh, xs1, feat, 0, 1);

    // Layer 1
    for (int d = 0; d < 2; ++d)
        gates_gemm_kernel<<<dim3(M / 16, 4), 128, 0, stream>>>(
            xs1, wih1h + (size_t)d * G4 * 2 * HID, biases + (size_t)(2 + d) * G4,
            gates + (size_t)d * M * G4, 2 * HID, (2 * HID) / 32);
    lstm_seq_kernel<<<dim3(BATCH / 16, 2), 512, 0, stream>>>(
        gates, whhh + (size_t)2 * G4 * HID, nullptr, feat, 2 * HID, 0);

    // Classifier head
    head_kernel<<<BATCH, 32, 0, stream>>>(feat, fcw, fcb, out);
}
